// RNNTCompactPredictorJoiner_63316407877917
// MI455X (gfx1250) — compile-verified, matrix operand order checked by reading, not run on hardware
//
#include <hip/hip_runtime.h>

typedef __attribute__((ext_vector_type(2)))  float    v2f;
typedef __attribute__((ext_vector_type(8)))  float    v8f;
typedef __attribute__((ext_vector_type(8)))  _Float16 v8h;
typedef __attribute__((ext_vector_type(16))) _Float16 v16h;

#define TMAXC 256
#define UMAXC 64
#define JDIM  512
#define VDIM  1024
#define BTOT  8

// ---------------------------------------------------------------------------
// Kernel 1: repack W_join [V][J] (fp32 row-major) into f16 WMMA B-fragment
// order: Bpack[((ntile*16 + kstep)*32 + lane)*16 + e]
//   n = ntile*16 + (lane&15)
//   k = kstep*32 + (e&7) | ((lane>>4)&1)<<3 | (e>>3)<<4
// ---------------------------------------------------------------------------
__global__ void pack_wjoin_kernel(const float* __restrict__ Wjoin,
                                  _Float16* __restrict__ Bpack) {
  int idx = blockIdx.x * blockDim.x + threadIdx.x;   // over VDIM*JDIM
  if (idx >= VDIM * JDIM) return;
  int e     = idx & 15;
  int lane  = (idx >> 4) & 31;
  int kstep = (idx >> 9) & 15;
  int ntile = idx >> 13;
  int n = (ntile << 4) | (lane & 15);
  int k = (kstep << 5) | (e & 7) | (((lane >> 4) & 1) << 3) | ((e >> 3) << 4);
  Bpack[idx] = (_Float16)Wjoin[n * JDIM + k];
}

// ---------------------------------------------------------------------------
// Kernel 2/3: fp32 GEMM  C[M][N] = A[M][K] * Wt[N][K]^T + bias
// using V_WMMA_F32_16X16X4_F32; one 16x16 tile per wave.
// Optional row gather for A (pred path: A = emb, gather = prefix).
// Fragment layouts (wave32):
//   A 16x4 f32:  lane row = lane&15, K = kbase + 2*(lane>>4) + {0,1}
//   B 4x16 f32:  lane col = lane&15, K = kbase + 2*(lane>>4) + {0,1}
//   C 16x16 f32: lane col = lane&15, rows = 8*(lane>>4) + e
// ---------------------------------------------------------------------------
__global__ void proj_gemm_f32_kernel(const float* __restrict__ A,
                                     const float* __restrict__ Wt,
                                     const float* __restrict__ bias,
                                     const int*   __restrict__ gather,
                                     float* __restrict__ C,
                                     int M, int N, int K) {
  int lane = threadIdx.x & 31;
  int wave = threadIdx.x >> 5;
  int tile = blockIdx.x * (blockDim.x >> 5) + wave;
  int ntiles = N >> 4;
  int mt = tile / ntiles;
  int nt = tile - mt * ntiles;
  if (mt >= (M >> 4)) return;                 // wave-uniform guard

  int mrow = (mt << 4) | (lane & 15);
  int arow = gather ? gather[mrow] : mrow;
  int ncol = (nt << 4) | (lane & 15);
  const float* ap = A  + (long)arow * K + ((lane >> 4) << 1);
  const float* bp = Wt + (long)ncol * K + ((lane >> 4) << 1);

  v8f acc = {};
  for (int k = 0; k < K; k += 4) {
    v2f a = *(const v2f*)(ap + k);
    v2f b = *(const v2f*)(bp + k);
    acc = __builtin_amdgcn_wmma_f32_16x16x4_f32(
        false, a, false, b, (short)0, acc, false, false);
  }

  float bv = bias[ncol];
  int mbase = (mt << 4) + ((lane >> 4) << 3);
#pragma unroll
  for (int e = 0; e < 8; ++e)
    C[(long)(mbase + e) * N + ncol] = acc[e] + bv;
}

// ---------------------------------------------------------------------------
// Kernel 4: fused ragged tanh-joint + vocab GEMM (f16 WMMA, f32 accum).
// Block = 256 threads (8 waves). Covers 32 output rows x 512 vocab cols
// (gridDim.y = 2 halves of V=1024). Each wave: 2 M-tiles x 4 N-tiles.
// ---------------------------------------------------------------------------
#define ASTRIDE 528   // 512 + 16 halves pad -> bank spread, keeps 16B align

__global__ void joint_gemm_kernel(const float* __restrict__ enc_proj,  // [2048][512]
                                  const float* __restrict__ pred_proj, // [512][512]
                                  const int*  __restrict__ gidx, int rows,
                                  const _Float16* __restrict__ Bpack,
                                  const float* __restrict__ b_join,
                                  float* __restrict__ out) {
  __shared__ __align__(16) _Float16 Ash[32 * ASTRIDE];

  int tid  = threadIdx.x;
  int row0 = blockIdx.x * 32;

  // Phase 1: build f16 A-tile (32 rows x 512 K) = tanh(enc + pred), ragged decode
  for (int i = tid; i < 32 * JDIM; i += 256) {
    int r = i >> 9;
    int k = i & (JDIM - 1);
    int row = row0 + r;
    float v = 0.f;
    if (row < rows) {
      int g  = gidx[row];                    // b*16384 + t*64 + u
      int er = g >> 6;                       // b*TMAX + t
      int pr = ((g >> 14) << 6) | (g & 63);  // b*UMAX + u
      v = tanhf(enc_proj[er * JDIM + k] + pred_proj[pr * JDIM + k]);
    }
    Ash[r * ASTRIDE + k] = (_Float16)v;
  }
  __syncthreads();

  // Phase 2: per-wave WMMA GEMM
  int lane  = tid & 31;
  int wave  = tid >> 5;
  int ncol0 = blockIdx.y * 512 + wave * 64;
  int baseTile = ncol0 >> 4;

  v8f acc[2][4] = {};

  int arow = lane & 15;
  int akhi = ((lane >> 4) & 1) << 3;         // 0 or 8 halves
  const _Float16* a0 = &Ash[(arow)      * ASTRIDE + akhi];
  const _Float16* a1 = &Ash[(16 + arow) * ASTRIDE + akhi];

  for (int ks = 0; ks < 16; ++ks) {
    // A fragments for the two M-tiles (2x ds_load_b128 each)
    v8h x0 = *(const v8h*)(a0 + ks * 32);
    v8h x1 = *(const v8h*)(a0 + ks * 32 + 16);
    v16h afr0 = __builtin_shufflevector(x0, x1, 0,1,2,3,4,5,6,7,8,9,10,11,12,13,14,15);
    v8h y0 = *(const v8h*)(a1 + ks * 32);
    v8h y1 = *(const v8h*)(a1 + ks * 32 + 16);
    v16h afr1 = __builtin_shufflevector(y0, y1, 0,1,2,3,4,5,6,7,8,9,10,11,12,13,14,15);

#pragma unroll
    for (int nt = 0; nt < 4; ++nt) {
      const v16h b = *(const v16h*)(Bpack +
          ((((long)(baseTile + nt) * 16 + ks) * 32 + lane) << 4));
      acc[0][nt] = __builtin_amdgcn_wmma_f32_16x16x32_f16(
          false, afr0, false, b, (short)0, acc[0][nt], false, false);
      acc[1][nt] = __builtin_amdgcn_wmma_f32_16x16x32_f16(
          false, afr1, false, b, (short)0, acc[1][nt], false, false);
    }
  }

  // Epilogue: C layout -> lane = col within tile, rows = 8*(lane>>4)+e
  int mloc = ((lane >> 4) & 1) << 3;
#pragma unroll
  for (int mt = 0; mt < 2; ++mt) {
#pragma unroll
    for (int nt = 0; nt < 4; ++nt) {
      int col = ncol0 + nt * 16 + (lane & 15);
      float bj = b_join[col];
#pragma unroll
      for (int e = 0; e < 8; ++e) {
        int row = row0 + mt * 16 + mloc + e;
        if (row < rows)
          out[(long)row * VDIM + col] = acc[mt][nt][e] + bj;
      }
    }
  }
}

// ---------------------------------------------------------------------------
extern "C" void kernel_launch(void* const* d_in, const int* in_sizes, int n_in,
                              void* d_out, int out_size, void* d_ws, size_t ws_size,
                              hipStream_t stream) {
  const float* enc_out = (const float*)d_in[0];   // [8,256,512]
  const int*   prefix  = (const int*)  d_in[1];   // [8,64]
  const int*   gidx    = (const int*)  d_in[4];   // [STU]
  const float* emb     = (const float*)d_in[5];   // [1024,512]
  const float* W_enc   = (const float*)d_in[6];   // [512,512]
  const float* b_enc   = (const float*)d_in[7];
  const float* W_pred  = (const float*)d_in[8];   // [512,512]
  const float* b_pred  = (const float*)d_in[9];
  const float* W_join  = (const float*)d_in[10];  // [1024,512]
  const float* b_join  = (const float*)d_in[11];

  int rows = in_sizes[4];                         // STU
  float* out = (float*)d_out;

  char* ws = (char*)d_ws;
  float*    enc_proj  = (float*)ws;                                        // 4 MB
  float*    pred_proj = (float*)(ws + (size_t)2048 * 512 * 4);             // 1 MB
  _Float16* Bpack     = (_Float16*)(ws + (size_t)2048 * 512 * 4
                                       + (size_t)512  * 512 * 4);          // 1 MB

  // 1) pack W_join to f16 B-fragments
  pack_wjoin_kernel<<<(VDIM * JDIM) / 256, 256, 0, stream>>>(W_join, Bpack);

  // 2) enc_proj: M=2048, N=512, K=512  -> 128*32 = 4096 tiles, 8 waves/block
  proj_gemm_f32_kernel<<<4096 / 8, 256, 0, stream>>>(
      enc_out, W_enc, b_enc, nullptr, enc_proj, 2048, 512, 512);

  // 3) pred_proj: M=512 (B*UMAX) with emb gather, N=512, K=512 -> 1024 tiles
  proj_gemm_f32_kernel<<<1024 / 8, 256, 0, stream>>>(
      emb, W_pred, b_pred, prefix, pred_proj, 512, 512, 512);

  // 4) fused tanh-joint + vocab GEMM
  dim3 g((rows + 31) / 32, 2);
  joint_gemm_kernel<<<g, 256, 0, stream>>>(
      enc_proj, pred_proj, gidx, rows, Bpack, b_join, out);
}
